// Moe_79018808312213
// MI455X (gfx1250) — matrix operand order repaired and checked by reading, not compiled, not run on hardware
//
#include <hip/hip_runtime.h>
#include <hip/hip_bf16.h>

typedef __bf16 bf16_t;
typedef __attribute__((ext_vector_type(16))) __bf16 v16bf;
typedef __attribute__((ext_vector_type(8)))  float  v8f;
typedef __attribute__((ext_vector_type(4)))  int    v4i;

#define AS1 __attribute__((address_space(1)))
#define AS3 __attribute__((address_space(3)))

#if defined(__has_builtin)
#if __has_builtin(__builtin_amdgcn_global_load_async_to_lds_b128) && \
    __has_builtin(__builtin_amdgcn_s_wait_asynccnt)
#define USE_ASYNC_LDS 1
#endif
#endif

static constexpr int T = 4096;   // B*S tokens
static constexpr int H = 2048;
static constexpr int F = 7168;
static constexpr int E = 8;
static constexpr int LDA = 40;   // LDS row stride (bf16) -> conflict-free b128 frag reads

// workspace layout (bytes); every size is a multiple of 1024
static constexpr size_t OFF_COUNTS = 0;                                    // E ints
static constexpr size_t OFF_TOK    = 1024;                                 // E*T ints
static constexpr size_t OFF_WTS    = OFF_TOK  + (size_t)E * T * 4;         // E*T floats
static constexpr size_t OFF_XBF    = OFF_WTS  + (size_t)E * T * 4;         // T x H bf16
static constexpr size_t OFF_W1BF   = OFF_XBF  + (size_t)T * H * 2;         // E x F x H bf16
static constexpr size_t OFF_W2BF   = OFF_W1BF + (size_t)E * F * H * 2;     // E x H x F bf16
static constexpr size_t OFF_W3BF   = OFF_W2BF + (size_t)E * H * F * 2;     // E x F x H bf16
static constexpr size_t OFF_H      = OFF_W3BF + (size_t)E * F * H * 2;     // 2T x F bf16
static constexpr size_t OFF_PART   = OFF_H    + (size_t)2 * T * F * 2;     // 2T x H f32

__device__ __forceinline__ v8f wmma_bf16(v16bf a, v16bf b, v8f c) {
  return __builtin_amdgcn_wmma_f32_16x16x32_bf16(false, a, false, b, (short)0, c,
                                                 false, false);
}

// convert 8 consecutive fp32 -> 8 bf16 and store 16B
__device__ __forceinline__ void cvt8(const float* __restrict__ src, bf16_t* dst) {
  const float4 a = reinterpret_cast<const float4*>(src)[0];
  const float4 b = reinterpret_cast<const float4*>(src)[1];
  union { uint4 q; __bf16 h[8]; } r;
  r.h[0] = (__bf16)a.x; r.h[1] = (__bf16)a.y; r.h[2] = (__bf16)a.z; r.h[3] = (__bf16)a.w;
  r.h[4] = (__bf16)b.x; r.h[5] = (__bf16)b.y; r.h[6] = (__bf16)b.z; r.h[7] = (__bf16)b.w;
  *reinterpret_cast<uint4*>(dst) = r.q;
}

// 16B global->LDS copy; async DMA path when the toolchain exposes it
__device__ __forceinline__ void copy16_g2l(const bf16_t* gsrc, bf16_t* ldst) {
#ifdef USE_ASYNC_LDS
  __builtin_amdgcn_global_load_async_to_lds_b128((AS1 v4i*)gsrc, (AS3 v4i*)ldst, 0, 0);
#else
  *reinterpret_cast<uint4*>(ldst) = *reinterpret_cast<const uint4*>(gsrc);
#endif
}

__device__ __forceinline__ void g2l_fence() {
#ifdef USE_ASYNC_LDS
  __builtin_amdgcn_s_wait_asynccnt(0);
#endif
}

// A fragment (16x32 bf16) from LDS: lane = row (lane&15); K = kb..kb+7, kb+16..kb+23
__device__ __forceinline__ v16bf afrag_lds(const bf16_t* base, int frag, int lane) {
  const int row  = frag * 16 + (lane & 15);
  const int koff = (lane & 16) >> 1;
  const bf16_t* p = base + row * LDA + koff;
  union { v16bf v; uint4 q[2]; } r;
  r.q[0] = *reinterpret_cast<const uint4*>(p);
  r.q[1] = *reinterpret_cast<const uint4*>(p + 16);
  return r.v;
}

// B fragment (32x16 bf16) from LDS: lane = column N (lane&15); K = kb..kb+7, kb+16..kb+23
__device__ __forceinline__ v16bf bfrag_lds(const bf16_t* base, int frag, int lane) {
  const int row  = frag * 16 + (lane & 15);
  const int koff = (lane & 16) >> 1;
  const bf16_t* p = base + row * LDA + koff;
  union { v16bf v; uint4 q[2]; } r;
  r.q[0] = *reinterpret_cast<const uint4*>(p);
  r.q[1] = *reinterpret_cast<const uint4*>(p + 16);
  return r.v;
}

// ---------------- fp32 -> bf16 pre-conversion (once per launch) ----------------
__global__ __launch_bounds__(256) void cvt_kernel(
    const float* __restrict__ src, bf16_t* __restrict__ dst, size_t n8) {
  const size_t i = ((size_t)blockIdx.x * blockDim.x + threadIdx.x);
  if (i < n8) cvt8(src + i * 8, dst + i * 8);
}

// ---------------- gating: one wave per token ----------------
__global__ __launch_bounds__(256) void gate_kernel(
    const float* __restrict__ x, const float* __restrict__ gw,
    int* __restrict__ counts, int* __restrict__ tok_slot, float* __restrict__ wts) {
  const int wave = threadIdx.x >> 5;
  const int lane = threadIdx.x & 31;
  const int t = blockIdx.x * 8 + wave;
  if (t >= T) return;
  const float* xr = x + (size_t)t * H;
  float acc[E];
#pragma unroll
  for (int e = 0; e < E; ++e) acc[e] = 0.f;
  for (int k = lane; k < H; k += 32) {
    const float xv = xr[k];
#pragma unroll
    for (int e = 0; e < E; ++e) acc[e] += xv * gw[e * H + k];
  }
#pragma unroll
  for (int e = 0; e < E; ++e) {
#pragma unroll
    for (int off = 16; off > 0; off >>= 1) acc[e] += __shfl_xor(acc[e], off, 32);
  }
  if (lane == 0) {
    int i0 = 0;
#pragma unroll
    for (int e = 1; e < E; ++e) if (acc[e] > acc[i0]) i0 = e;
    int i1 = (i0 == 0) ? 1 : 0;
#pragma unroll
    for (int e = 0; e < E; ++e) if (e != i0 && acc[e] > acc[i1]) i1 = e;
    const float d  = __expf(acc[i1] - acc[i0]);   // normalized top-2 softmax
    const float w0 = __builtin_amdgcn_rcpf(1.f + d);
    const float w1 = d * w0;
    int p0 = atomicAdd(&counts[i0], 1);
    tok_slot[i0 * T + p0] = (t << 1);
    wts[i0 * T + p0] = w0;
    int p1 = atomicAdd(&counts[i1], 1);
    tok_slot[i1 * T + p1] = (t << 1) | 1;
    wts[i1 * T + p1] = w1;
  }
}

// ---------------- pass 1: h = silu(x@w1^T) * (x@w3^T) ----------------
// block tile: M=128 entries x N=128 F-cols; 16 waves = 4(M) x 4(N); wave = 2x2 frags x {w1,w3}
__global__ __launch_bounds__(512) void ffn1_kernel(
    const bf16_t* __restrict__ xbf, const bf16_t* __restrict__ w1bf,
    const bf16_t* __restrict__ w3bf,
    const int* __restrict__ counts, const int* __restrict__ tok_slot,
    bf16_t* __restrict__ hbuf) {
  __shared__ bf16_t As[128 * LDA];
  __shared__ bf16_t B1s[128 * LDA];
  __shared__ bf16_t B3s[128 * LDA];

  const int e = blockIdx.z;
  const int count = counts[e];
  const int m0 = blockIdx.y * 128;
  if (m0 >= count) return;
  const int tid  = threadIdx.x;
  const int lane = tid & 31;
  const int wave = tid >> 5;
  const int n_base = blockIdx.x * 128;

  // staging: A rows gathered per token, B rows streamed; 16B per copy
  const int arow = tid >> 2;              // 0..127
  const int acg  = (tid & 3) * 8;         // 0,8,16,24
  const int aentry = min(m0 + arow, count - 1);
  const bf16_t* a_src = xbf + (size_t)(tok_slot[e * T + aentry] >> 1) * H + acg;
  bf16_t* a_dst = As + arow * LDA + acg;

  const int half = tid >> 8;              // 0 -> w1, 1 -> w3
  const int bt   = tid & 255;
  const int brow = bt >> 1;               // 0..127
  const int bcg  = (bt & 1) * 16;         // 0,16
  const bf16_t* b_src = (half ? w3bf : w1bf) + ((size_t)e * F + n_base + brow) * H + bcg;
  bf16_t* b_dst = (half ? B3s : B1s) + brow * LDA + bcg;

  const int mf0 = (wave >> 2) * 2;        // 0,2,4,6
  const int nf0 = (wave & 3) * 2;         // 0,2,4,6

  v8f g00 = {}, g01 = {}, g10 = {}, g11 = {};
  v8f u00 = {}, u01 = {}, u10 = {}, u11 = {};

  for (int k0 = 0; k0 < H; k0 += 32) {
    copy16_g2l(a_src + k0, a_dst);
    copy16_g2l(b_src + k0,     b_dst);
    copy16_g2l(b_src + k0 + 8, b_dst + 8);
    if (k0 + 128 < H) __builtin_prefetch(b_src + k0 + 128, 0, 3);
    g2l_fence();
    __syncthreads();
    const v16bf a0 = afrag_lds(As, mf0,     lane);
    const v16bf a1 = afrag_lds(As, mf0 + 1, lane);
    const v16bf p0 = bfrag_lds(B1s, nf0,     lane);
    const v16bf p1 = bfrag_lds(B1s, nf0 + 1, lane);
    const v16bf q0 = bfrag_lds(B3s, nf0,     lane);
    const v16bf q1 = bfrag_lds(B3s, nf0 + 1, lane);
    g00 = wmma_bf16(a0, p0, g00);
    g01 = wmma_bf16(a0, p1, g01);
    g10 = wmma_bf16(a1, p0, g10);
    g11 = wmma_bf16(a1, p1, g11);
    u00 = wmma_bf16(a0, q0, u00);
    u01 = wmma_bf16(a0, q1, u01);
    u10 = wmma_bf16(a1, q0, u10);
    u11 = wmma_bf16(a1, q1, u11);
    __syncthreads();
  }

  const int ncol0 = n_base + nf0 * 16 + (lane & 15);
  const int mhi   = (lane & 16) >> 1;
  v8f* G[2][2] = {{&g00, &g01}, {&g10, &g11}};
  v8f* U[2][2] = {{&u00, &u01}, {&u10, &u11}};
#pragma unroll
  for (int mi = 0; mi < 2; ++mi) {
#pragma unroll
    for (int v = 0; v < 8; ++v) {
      const int entry = min(m0 + (mf0 + mi) * 16 + v + mhi, count - 1);
      const size_t hrow = (size_t)tok_slot[e * T + entry] * F;
#pragma unroll
      for (int ni = 0; ni < 2; ++ni) {
        const float gv = (*G[mi][ni])[v];
        const float hv = gv * __builtin_amdgcn_rcpf(1.f + __expf(-gv)) * (*U[mi][ni])[v];
        hbuf[hrow + ncol0 + ni * 16] = (bf16_t)hv;
      }
    }
  }
}

// ---------------- pass 2: partial = weight * (h @ w2^T) ----------------
// block tile: M=128 entries x N=128 H-cols; 16 waves = 4(M) x 4(N)
__global__ __launch_bounds__(512) void ffn2_kernel(
    const bf16_t* __restrict__ hbuf, const bf16_t* __restrict__ w2bf,
    const int* __restrict__ counts, const int* __restrict__ tok_slot,
    const float* __restrict__ wts, float* __restrict__ part) {
  __shared__ bf16_t As[128 * LDA];
  __shared__ bf16_t Bs[128 * LDA];

  const int e = blockIdx.z;
  const int count = counts[e];
  const int m0 = blockIdx.y * 128;
  if (m0 >= count) return;
  const int tid  = threadIdx.x;
  const int lane = tid & 31;
  const int wave = tid >> 5;
  const int n_base = blockIdx.x * 128;

  const int arow = tid >> 2;              // 0..127
  const int acg  = (tid & 3) * 8;
  const int aentry = min(m0 + arow, count - 1);
  const bf16_t* a_src = hbuf + (size_t)tok_slot[e * T + aentry] * F + acg;
  bf16_t* a_dst = As + arow * LDA + acg;

  const int brow = tid >> 2;              // 0..127
  const int bcg  = (tid & 3) * 8;
  const bf16_t* b_src = w2bf + ((size_t)e * H + n_base + brow) * F + bcg;
  bf16_t* b_dst = Bs + brow * LDA + bcg;

  const int mf0 = (wave >> 2) * 2;
  const int nf0 = (wave & 3) * 2;

  v8f c00 = {}, c01 = {}, c10 = {}, c11 = {};

  for (int k0 = 0; k0 < F; k0 += 32) {
    copy16_g2l(a_src + k0, a_dst);
    copy16_g2l(b_src + k0, b_dst);
    if (k0 + 128 < F) __builtin_prefetch(b_src + k0 + 128, 0, 3);
    g2l_fence();
    __syncthreads();
    const v16bf a0 = afrag_lds(As, mf0,     lane);
    const v16bf a1 = afrag_lds(As, mf0 + 1, lane);
    const v16bf b0 = bfrag_lds(Bs, nf0,     lane);
    const v16bf b1 = bfrag_lds(Bs, nf0 + 1, lane);
    c00 = wmma_bf16(a0, b0, c00);
    c01 = wmma_bf16(a0, b1, c01);
    c10 = wmma_bf16(a1, b0, c10);
    c11 = wmma_bf16(a1, b1, c11);
    __syncthreads();
  }

  const int ncol0 = n_base + nf0 * 16 + (lane & 15);
  const int mhi   = (lane & 16) >> 1;
  v8f* C[2][2] = {{&c00, &c01}, {&c10, &c11}};
#pragma unroll
  for (int mi = 0; mi < 2; ++mi) {
#pragma unroll
    for (int v = 0; v < 8; ++v) {
      const int entry = min(m0 + (mf0 + mi) * 16 + v + mhi, count - 1);
      const size_t prow = (size_t)tok_slot[e * T + entry] * H;
      const float w = wts[e * T + entry];
#pragma unroll
      for (int ni = 0; ni < 2; ++ni) {
        part[prow + ncol0 + ni * 16] = w * (*C[mi][ni])[v];
      }
    }
  }
}

// ---------------- combine the two slots per token ----------------
__global__ __launch_bounds__(256) void combine_kernel(
    const float* __restrict__ part, float* __restrict__ out) {
  const size_t idx = (size_t)blockIdx.x * blockDim.x + threadIdx.x;
  const size_t t = idx / H;
  const size_t h = idx % H;
  out[idx] = part[(2 * t) * H + h] + part[(2 * t + 1) * H + h];
}

extern "C" void kernel_launch(void* const* d_in, const int* in_sizes, int n_in,
                              void* d_out, int out_size, void* d_ws, size_t ws_size,
                              hipStream_t stream) {
  const float* x  = (const float*)d_in[0];
  const float* gw = (const float*)d_in[1];
  const float* w1 = (const float*)d_in[2];
  const float* w2 = (const float*)d_in[3];
  const float* w3 = (const float*)d_in[4];
  float* out = (float*)d_out;

  char* ws = (char*)d_ws;
  int*    counts   = (int*)(ws + OFF_COUNTS);
  int*    tok_slot = (int*)(ws + OFF_TOK);
  float*  wts      = (float*)(ws + OFF_WTS);
  bf16_t* xbf      = (bf16_t*)(ws + OFF_XBF);
  bf16_t* w1bf     = (bf16_t*)(ws + OFF_W1BF);
  bf16_t* w2bf     = (bf16_t*)(ws + OFF_W2BF);
  bf16_t* w3bf     = (bf16_t*)(ws + OFF_W3BF);
  bf16_t* hbuf     = (bf16_t*)(ws + OFF_H);
  float*  part     = (float*)(ws + OFF_PART);

  (void)hipMemsetAsync(counts, 0, 1024, stream);

  const size_t nx = (size_t)T * H / 8;         // vec8 elements
  const size_t nw = (size_t)E * F * H / 8;
  cvt_kernel<<<(nx + 255) / 256, 256, 0, stream>>>(x,  xbf,  nx);
  cvt_kernel<<<(nw + 255) / 256, 256, 0, stream>>>(w1, w1bf, nw);
  cvt_kernel<<<(nw + 255) / 256, 256, 0, stream>>>(w2, w2bf, nw);
  cvt_kernel<<<(nw + 255) / 256, 256, 0, stream>>>(w3, w3bf, nw);

  gate_kernel<<<T / 8, 256, 0, stream>>>(x, gw, counts, tok_slot, wts);
  ffn1_kernel<<<dim3(F / 128, T / 128, E), 512, 0, stream>>>(xbf, w1bf, w3bf, counts,
                                                             tok_slot, hbuf);
  ffn2_kernel<<<dim3(H / 128, T / 128, E), 512, 0, stream>>>(hbuf, w2bf, counts,
                                                             tok_slot, wts, part);
  combine_kernel<<<((size_t)T * H) / 256, 256, 0, stream>>>(part, out);
}
